// QuantizedLSTM_73684458930247
// MI455X (gfx1250) — compile-verified, matrix operand order checked by reading, not run on hardware
//
#include <hip/hip_runtime.h>

// ---------------------------------------------------------------------------
// Quantized 2-layer LSTM for MI455X (gfx1250, wave32, WMMA + Tensor Data Mover)
// gates[64,4096] = z[64,2048] @ Wcat^T  (f16 WMMA, f32 accum), fused cell math.
// A-tiles are DMA'd into LDS by the TDM (tensor_load_to_lds) with HW padding
// for a conflict-free 136-halfword row stride, double-buffered against WMMA.
// ---------------------------------------------------------------------------

typedef __attribute__((ext_vector_type(16))) _Float16 v16h;
typedef __attribute__((ext_vector_type(8)))  _Float16 v8h;
typedef __attribute__((ext_vector_type(8)))  float    v8f;
typedef unsigned int u32x4 __attribute__((ext_vector_type(4)));
typedef int          i32x8 __attribute__((ext_vector_type(8)));
typedef int          i32x4 __attribute__((ext_vector_type(4)));

#define BB    64          // batch
#define SS    512         // sequence
#define DD    1024        // input dim
#define HH    1024        // hidden
#define KTOT  2048        // D + H (concat GEMM K)
#define NGATE 4096        // 4*H gate rows
#define BK    128         // K block staged in LDS
#define LDA   136         // padded LDS row stride (f16): 272B -> conflict-free

#if __has_builtin(__builtin_amdgcn_tensor_load_to_lds)
#define HAVE_TDM 1
#else
#define HAVE_TDM 0
#endif

// TF32-like quantization: 10 mantissa bits, 8 exponent bits, RNE-ish.
__device__ __forceinline__ float quant10(float x) {
  float ax = fabsf(x);
  float e  = floorf(log2f(fmaxf(ax, 1e-38f)));
  e = fminf(fmaxf(e, -126.f), 127.f);
  float s  = exp2f(10.f - e);
  float q  = rintf(x * s) / s;
  float mv = (2.0f - exp2f(-10.f)) * exp2f(127.f);
  q = fminf(fmaxf(q, -mv), mv);
  return (ax == 0.f) ? 0.f : q;
}

__device__ __forceinline__ float sigm(float x) {
  return 1.0f / (1.0f + expf(-x));
}

// Build a 16x16x32 f16 fragment from two 16B chunks (K+0..7 / K+16..23 per
// the CDNA5 16-bit A/B VGPR layout; caller applies the lane>=16 "+8" offset).
__device__ __forceinline__ v16h ldfrag(const _Float16* p) {
  v8h lo = *(const v8h*)(p);
  v8h hi = *(const v8h*)(p + 16);
  return __builtin_shufflevector(lo, hi, 0,1,2,3,4,5,6,7,8,9,10,11,12,13,14,15);
}

#if HAVE_TDM
// Issue a TDM load of a 64-row x 128-col f16 tile (row stride = stride_elems)
// into LDS at lds_byte_addr, with HW padding of 4 DWORDs every 64 DWORDs so
// the LDS row stride becomes 272B (== LDA halfwords). D# per ISA ch.8.
__device__ __forceinline__ void tdm_load_tile(const _Float16* src,
                                              long stride_elems,
                                              unsigned lds_byte_addr) {
  unsigned long long ga = (unsigned long long)(uintptr_t)src;
  u32x4 g0 = { 1u,                                  // count=1 valid descriptor
               lds_byte_addr,                       // lds_addr
               (unsigned)ga,                        // global_addr[31:0]
               ((unsigned)(ga >> 32) & 0x01FFFFFFu) // global_addr[56:32]
                 | (2u << 30) };                    // type=2 ("image")
  // group1: data_size=2B, pad_enable, pad_interval=64DW(code5), pad_amount=4DW(code3)
  unsigned w0 = (1u << 16) | (1u << 20) | (5u << 22) | (3u << 25);
  i32x8 g1 = { (int)w0,
               0,                                   // atomic barrier addr
               0x4000,                              // tensor_dim0 = 1<<30 (hi16)
               (int)(0x4000u | (128u << 16)),       // tensor_dim1 hi16; tile_dim0=128
               64,                                  // tile_dim1 = 64 rows
               (int)(unsigned)stride_elems,         // tensor_dim0_stride lo32
               0, 0 };
  i32x4 z4 = { 0, 0, 0, 0 };
#if defined(__clang_major__) && __clang_major__ >= 23
  i32x8 z8 = { 0, 0, 0, 0, 0, 0, 0, 0 };
  __builtin_amdgcn_tensor_load_to_lds(g0, g1, z4, z4, z8, 0);
#else
  __builtin_amdgcn_tensor_load_to_lds(g0, g1, z4, z4, 0);
#endif
}

__device__ __forceinline__ void tdm_wait() {
#if __has_builtin(__builtin_amdgcn_s_wait_tensorcnt)
  __builtin_amdgcn_s_wait_tensorcnt(0);
#else
  asm volatile("s_wait_tensorcnt 0x0" ::: "memory");
#endif
}
#endif  // HAVE_TDM

// One K-block of WMMAs: 4 k-steps x 4 gates = 16 WMMAs.
__device__ __forceinline__ void gemm_block(const _Float16* sAb, int arow, int koff,
                                           const _Float16* w_i, const _Float16* w_f,
                                           const _Float16* w_g, const _Float16* w_o,
                                           int kb, v8f& ai, v8f& af, v8f& ag, v8f& ao) {
  #pragma unroll
  for (int ks = 0; ks < BK; ks += 32) {
    v16h a  = ldfrag(&sAb[arow * LDA + ks + koff]);
    v16h bi = ldfrag(w_i + kb + ks);
    v16h bf = ldfrag(w_f + kb + ks);
    v16h bg = ldfrag(w_g + kb + ks);
    v16h bo = ldfrag(w_o + kb + ks);
    ai = __builtin_amdgcn_wmma_f32_16x16x32_f16(false, a, false, bi, (short)0, ai, false, false);
    af = __builtin_amdgcn_wmma_f32_16x16x32_f16(false, a, false, bf, (short)0, af, false, false);
    ag = __builtin_amdgcn_wmma_f32_16x16x32_f16(false, a, false, bg, (short)0, ag, false, false);
    ao = __builtin_amdgcn_wmma_f32_16x16x32_f16(false, a, false, bo, (short)0, ao, false, false);
  }
}

// ---------------- one-time packing kernels ----------------

__global__ __launch_bounds__(256) void pack_w_kernel(const float* __restrict__ wih,
                                                     const float* __restrict__ whh,
                                                     _Float16* __restrict__ wcat) {
  size_t idx = (size_t)blockIdx.x * 256 + threadIdx.x;   // 4096*2048 total
  int row = (int)(idx >> 11);
  int col = (int)(idx & 2047);
  float v = (col < 1024) ? wih[(size_t)row * 1024 + col]
                         : whh[(size_t)row * 1024 + (col - 1024)];
  wcat[idx] = (_Float16)quant10(v);
}

__global__ __launch_bounds__(256) void pack_x_kernel(const float* __restrict__ x,
                                                     _Float16* __restrict__ qx) {
  size_t i = (size_t)blockIdx.x * 256 + threadIdx.x;     // B*S*D total
  qx[i] = (_Float16)quant10(x[i]);
}

__global__ __launch_bounds__(256) void pack_b_kernel(const float* __restrict__ bih,
                                                     const float* __restrict__ bhh,
                                                     float* __restrict__ bsum) {
  int i = blockIdx.x * 256 + threadIdx.x;                // 4096 total
  bsum[i] = quant10(bih[i]) + quant10(bhh[i]);
}

__global__ __launch_bounds__(256) void zero_kernel(float* __restrict__ p, long n) {
  long i = (long)blockIdx.x * 256 + threadIdx.x;
  if (i < n) p[i] = 0.f;
}

__global__ __launch_bounds__(256) void tail_copy_kernel(const float* __restrict__ hsrc,
                                                        const float* __restrict__ csrc,
                                                        float* __restrict__ dst, long n) {
  long i = (long)blockIdx.x * 256 + threadIdx.x;
  if (i < n)            dst[i] = hsrc[i];
  else if (i < 2 * n)   dst[i] = csrc[i - n];
}

// ---------------- fused LSTM layer-step kernel ----------------
//
// Grid: 32 WGs x 256 threads (8 wave32).  WG b owns h-columns [b*32, b*32+32)
// across ALL FOUR gates so the cell update is fully in-register.
// Wave w: m_tile = w%4, hcol_tile = w/4; 4 accumulators (i,f,g,o) per wave.
// A (z = [inp16 | h16prev]) is DMA'd into LDS by the TDM, double buffered.
__global__ __launch_bounds__(256) void lstm_step_kernel(
    const _Float16* __restrict__ inp16, long inp_stride,   // f16 input rows
    const _Float16* __restrict__ hprev16,                  // f16 [64][1024]
    const _Float16* __restrict__ wcat,                     // [4096][2048] f16
    const float* __restrict__ bsum,                        // [4096] f32
    float* __restrict__ cstate,                            // [64][1024] in-place
    float* __restrict__ hnew,                              // [64][1024] f32
    _Float16* __restrict__ h16new,                         // [64][1024] f16
    float* __restrict__ outp, long out_stride)             // nullable
{
  __shared__ _Float16 sA[2][BB * LDA];

  const int tid  = threadIdx.x;
  const int lane = tid & 31;
  const int wv   = tid >> 5;
  const int m0   = (wv & 3) * 16;
  const int ncol = blockIdx.x * 32 + (wv >> 2) * 16 + (lane & 15); // h column
  const int koff = (lane & 16) ? 8 : 0;
  const int arow = m0 + (lane & 15);

  v8f acc_i = {}; v8f acc_f = {}; v8f acc_g = {}; v8f acc_o = {};

  const _Float16* w_i = wcat + (size_t)(0 * HH + ncol) * KTOT + koff;
  const _Float16* w_f = wcat + (size_t)(1 * HH + ncol) * KTOT + koff;
  const _Float16* w_g = wcat + (size_t)(2 * HH + ncol) * KTOT + koff;
  const _Float16* w_o = wcat + (size_t)(3 * HH + ncol) * KTOT + koff;

#if HAVE_TDM
  // prologue: DMA block kb=0 into buffer 0
  if (wv == 0) {
    tdm_load_tile(inp16, inp_stride, (unsigned)(uintptr_t)&sA[0][0]);
    tdm_wait();
  }
  __syncthreads();

  int cur = 0;
  for (int kb = 0; kb < KTOT; kb += BK) {
    // overlap: issue next block's DMA into the other buffer while computing
    if (wv == 0 && kb + BK < KTOT) {
      int knx = kb + BK;
      const _Float16* src = (knx < DD) ? (inp16 + knx) : (hprev16 + (knx - DD));
      long stride = (knx < DD) ? inp_stride : (long)HH;
      tdm_load_tile(src, stride, (unsigned)(uintptr_t)&sA[1 - cur][0]);
    }
    if (kb + BK < KTOT) {
      __builtin_prefetch(w_i + kb + BK, 0, 0);
      __builtin_prefetch(w_f + kb + BK, 0, 0);
      __builtin_prefetch(w_g + kb + BK, 0, 0);
      __builtin_prefetch(w_o + kb + BK, 0, 0);
    }
    gemm_block(&sA[cur][0], arow, koff, w_i, w_f, w_g, w_o, kb,
               acc_i, acc_f, acc_g, acc_o);
    if (wv == 0) tdm_wait();     // next buffer's DMA complete (TENSORcnt)
    __syncthreads();             // everyone done with cur; next data visible
    cur ^= 1;
  }
#else
  // fallback: cooperative f16 copy to LDS (no TDM builtin on this toolchain)
  for (int kb = 0; kb < KTOT; kb += BK) {
    __syncthreads();
    const _Float16* src = (kb < DD) ? (inp16 + kb) : (hprev16 + (kb - DD));
    long sstride = (kb < DD) ? inp_stride : (long)HH;
    #pragma unroll
    for (int it = 0; it < 16; ++it) {
      int pidx = it * 256 + tid;         // 4096 u32 = 64*128 f16
      int r    = pidx >> 6;
      int cp   = (pidx & 63) * 2;
      unsigned v = *(const unsigned*)(src + (size_t)r * sstride + cp);
      *(unsigned*)(&sA[0][r * LDA + cp]) = v;
    }
    __syncthreads();
    gemm_block(&sA[0][0], arow, koff, w_i, w_f, w_g, w_o, kb,
               acc_i, acc_f, acc_g, acc_o);
  }
#endif

  // ---- fused LSTM cell update (biases, activations, c/h) ----
  const float bi = bsum[0 * HH + ncol];
  const float bf = bsum[1 * HH + ncol];
  const float bg = bsum[2 * HH + ncol];
  const float bo = bsum[3 * HH + ncol];
  const int rbase = m0 + ((lane >> 4) << 3);   // accumulator row -> batch row

  #pragma unroll
  for (int r = 0; r < 8; ++r) {
    int m = rbase + r;
    size_t off = (size_t)m * HH + ncol;
    float iv = sigm(acc_i[r] + bi);
    float fv = sigm(acc_f[r] + bf);
    float gv = tanhf(acc_g[r] + bg);
    float ov = sigm(acc_o[r] + bo);
    float cn = fv * cstate[off] + iv * gv;
    float hn = ov * tanhf(cn);
    cstate[off] = cn;
    hnew[off]   = hn;
    h16new[off] = (_Float16)hn;
    if (outp) outp[(size_t)m * out_stride + ncol] = hn;
  }
}

// ---------------- host orchestration ----------------

extern "C" void kernel_launch(void* const* d_in, const int* in_sizes, int n_in,
                              void* d_out, int out_size, void* d_ws, size_t ws_size,
                              hipStream_t stream) {
  const float* x    = (const float*)d_in[0];
  const float* wih0 = (const float*)d_in[1];
  const float* whh0 = (const float*)d_in[2];
  const float* bih0 = (const float*)d_in[3];
  const float* bhh0 = (const float*)d_in[4];
  const float* wih1 = (const float*)d_in[5];
  const float* whh1 = (const float*)d_in[6];
  const float* bih1 = (const float*)d_in[7];
  const float* bhh1 = (const float*)d_in[8];
  float* out = (float*)d_out;

  const size_t WEL  = (size_t)NGATE * KTOT;       // f16 elems per layer (8M)
  const size_t XEL  = (size_t)BB * SS * DD;       // f16 elems of qx (32M)
  const long   HS   = (long)BB * HH;              // 65536
  const long   HS2  = 2 * HS;

  _Float16* wcat = (_Float16*)d_ws;               // 2*WEL f16   (32 MB)
  _Float16* qx16 = wcat + 2 * WEL;                // XEL f16     (64 MB)
  float*    bsum = (float*)(qx16 + XEL);          // 2*4096 f32
  float*    hbuf = bsum + 2 * NGATE;              // [2 pp][2 layer][64][1024] f32
  float*    cbuf = hbuf + 2 * HS2;                // [2 layer][64][1024] f32
  _Float16* h16  = (_Float16*)(cbuf + HS2);       // [2 pp][2 layer][64][1024] f16

  // one-time packing (deterministic, re-done every call)
  pack_w_kernel<<<(unsigned)(WEL / 256), 256, 0, stream>>>(wih0, whh0, wcat);
  pack_w_kernel<<<(unsigned)(WEL / 256), 256, 0, stream>>>(wih1, whh1, wcat + WEL);
  pack_x_kernel<<<(unsigned)(XEL / 256), 256, 0, stream>>>(x, qx16);
  pack_b_kernel<<<NGATE / 256, 256, 0, stream>>>(bih0, bhh0, bsum);
  pack_b_kernel<<<NGATE / 256, 256, 0, stream>>>(bih1, bhh1, bsum + NGATE);

  // zero hbuf + cbuf + h16 (contiguous: 2*HS2 + HS2 + HS2 floats worth)
  long zn = 4 * HS2;
  zero_kernel<<<(unsigned)((zn + 255) / 256), 256, 0, stream>>>(hbuf, zn);

  for (int t = 0; t < SS; ++t) {
    int pp = t & 1;
    float*    hnew  = hbuf + (size_t)(1 - pp) * HS2;
    _Float16* h16o  = h16  + (size_t)pp * HS2;        // prev f16 states
    _Float16* h16n  = h16  + (size_t)(1 - pp) * HS2;  // new  f16 states
    // layer 0: inp = qx16[:, t, :], h = h16o[0]
    lstm_step_kernel<<<32, 256, 0, stream>>>(
        qx16 + (size_t)t * DD, (long)SS * DD,
        h16o + 0 * HS, wcat, bsum,
        cbuf + 0 * HS, hnew + 0 * HS, h16n + 0 * HS, nullptr, 0);
    // layer 1: inp = fresh h16 of layer 0, h = h16o[1]; writes output[:, t, :]
    lstm_step_kernel<<<32, 256, 0, stream>>>(
        h16n + 0 * HS, (long)HH,
        h16o + 1 * HS, wcat + WEL, bsum + NGATE,
        cbuf + 1 * HS, hnew + 1 * HS, h16n + 1 * HS,
        out + (size_t)t * HH, (long)SS * HH);
  }

  // final (h, c) appended after output; S even -> final f32 state in hbuf pp=0
  float* tail = out + (size_t)BB * SS * HH;
  tail_copy_kernel<<<(unsigned)((2 * HS2 + 255) / 256), 256, 0, stream>>>(
      hbuf, cbuf, tail, HS2);
}
